// EdgeGeneration_47510928228699
// MI455X (gfx1250) — compile-verified, matrix operand order checked by reading, not run on hardware
//
#include <hip/hip_runtime.h>

#define N_NODES 200001
#define N_SUB   150000
#define FEAT    512
#define LABEL   64
#define ROWS    128
#define ASTRIDE 520   /* FEAT + 8 pad (bf16 elems): stride = 260 dwords -> 4-bank step, conflict-free b128 frag loads */
#define SSTRIDE 72    /* 64 + 8 pad */
#define GBLK    64

typedef __attribute__((ext_vector_type(16))) __bf16 bf16x16;
typedef __attribute__((ext_vector_type(8)))  __bf16 bf16x8;
typedef __attribute__((ext_vector_type(4)))  __bf16 bf16x4;
typedef __attribute__((ext_vector_type(8)))  float  f32x8;

__device__ __forceinline__ __bf16 tobf(float f) {
    unsigned u = __float_as_uint(f);
    unsigned short s = (unsigned short)((u + 0x7FFFu + ((u >> 16) & 1u)) >> 16);
    __bf16 b;
    __builtin_memcpy(&b, &s, 2);
    return b;
}

__device__ __forceinline__ bf16x16 mk_frag(const __bf16* lo, const __bf16* hi) {
    bf16x8 a = *(const bf16x8*)lo;
    bf16x8 b = *(const bf16x8*)hi;
    bf16x16 r;
#pragma unroll
    for (int i = 0; i < 8; ++i) { r[i] = a[i]; r[i + 8] = b[i]; }
    return r;
}

__device__ __forceinline__ f32x8 wmma_bf(bf16x16 a, bf16x16 b, f32x8 c) {
    return __builtin_amdgcn_wmma_f32_16x16x32_bf16(false, a, false, b, (short)0, c, false, false);
}

__device__ __forceinline__ float leaky(float x) { return x > 0.f ? x : 0.01f * x; }

// ---------------- prep kernels ----------------

// W12 = weight1(512x256) @ weight2(256x64); also store transposed bf16 [64][512]
extern "C" __global__ void prep_w12(const float* __restrict__ w1_, const float* __restrict__ w2_,
                                    float* __restrict__ W12, __bf16* __restrict__ W12T) {
    int idx = blockIdx.x * 256 + threadIdx.x;
    if (idx >= FEAT * LABEL) return;
    int i = idx >> 6, n = idx & 63;
    float acc = 0.f;
    for (int j = 0; j < 256; ++j) acc += w1_[i * 256 + j] * w2_[j * 64 + n];
    W12[idx] = acc;
    W12T[n * FEAT + i] = tobf(acc);
}

// W1sT[n][k] = W1[64+k][n] (bf16), W2T[n][k] = W2[k][n] (bf16)
extern "C" __global__ void prep_conv(const float* __restrict__ W1, const float* __restrict__ W2,
                                     __bf16* __restrict__ W1sT, __bf16* __restrict__ W2T) {
    int idx = blockIdx.x * 256 + threadIdx.x;
    if (idx < 512 * 64) {
        int n = idx >> 6, k = idx & 63;
        W1sT[n * 64 + k] = tobf(W1[(64 + k) * 512 + n]);
    }
    int idx2 = idx - 512 * 64;
    if (idx2 >= 0 && idx2 < 32 * 512) {
        int n = idx2 >> 9, k = idx2 & 511;
        W2T[n * 512 + k] = tobf(W2[k * 32 + n]);
    }
}

// c1[j] = b1[j] + tar_xw@W1[0:64] + add_xw@W1[128:192] + wlabel@W1[193:705] + wsec@W1[705:1217]
extern "C" __global__ void prep_const(const float* __restrict__ feat, const int* __restrict__ target,
                                      const float* __restrict__ W12, const float* __restrict__ wlabel,
                                      const float* __restrict__ wsec, const float* __restrict__ W1,
                                      const float* __restrict__ b1, float* __restrict__ c1) {
    __shared__ float txw[64], axw[64];
    int t = threadIdx.x;  // 512 threads
    if (t < 128) {
        int k = t & 63;
        int row = (t < 64) ? target[0] : (N_NODES - 1);
        const float* fr = feat + (size_t)row * FEAT;
        float acc = 0.f;
        for (int i = 0; i < FEAT; ++i) acc += fr[i] * W12[i * 64 + k];
        if (t < 64) txw[k] = acc; else axw[k] = acc;
    }
    __syncthreads();
    int j = t;
    float acc = b1[j];
    for (int k = 0; k < 64; ++k) acc += txw[k] * W1[k * 512 + j];
    for (int k = 0; k < 64; ++k) acc += axw[k] * W1[(128 + k) * 512 + j];
    for (int f = 0; f < 512; ++f) acc += wlabel[f] * W1[(193 + f) * 512 + j];
    for (int f = 0; f < 512; ++f) acc += wsec[f] * W1[(705 + f) * 512 + j];
    c1[j] = acc;
}

// zero score region, fill idx region, zero topk control block
extern "C" __global__ void init_out(const int* __restrict__ subn, float* __restrict__ out,
                                    unsigned* __restrict__ ctrl32) {
    int i = blockIdx.x * 256 + threadIdx.x;
    if (i < N_SUB) {
        out[i] = 0.f;
        out[N_SUB + i] = (float)subn[i];
    }
    if (i < 448) ctrl32[i] = 0u;
}

// ---------------- fused main: gather + 3 WMMA GEMMs + logits ----------------
extern "C" __global__ void __launch_bounds__(256)
fused_main(const int* __restrict__ subn, const float* __restrict__ feat,
           const float* __restrict__ adj, const __bf16* __restrict__ W12T,
           const __bf16* __restrict__ W1sT, const __bf16* __restrict__ W2T,
           const float* __restrict__ c1, const float* __restrict__ W1,
           const float* __restrict__ b2, const float* __restrict__ W3,
           const float* __restrict__ b3, float* __restrict__ logits) {
    extern __shared__ char smem[];
    __bf16* A   = (__bf16*)smem;                                      // ROWS*ASTRIDE bf16 (feat, then h1)
    __bf16* S   = (__bf16*)(smem + ROWS * ASTRIDE * 2);               // ROWS*SSTRIDE bf16 (sub_xw)
    float*  RED = (float*)(smem + ROWS * ASTRIDE * 2);                // overlaps S after GEMM2 done
    float*  ADJ = (float*)(smem + ROWS * ASTRIDE * 2 + ROWS * SSTRIDE * 2);

    const int tid = threadIdx.x;
    const int rowsBase = blockIdx.x * ROWS;

    // ---- Stage 1: gather feature rows, convert f32->bf16 into LDS ----
    for (int idx = tid; idx < ROWS * (FEAT / 4); idx += 256) {
        int r  = idx >> 7;
        int c4 = idx & 127;
        int grow = rowsBase + r; if (grow > N_SUB - 1) grow = N_SUB - 1;
        int src = subn[grow];
        const float4 v = ((const float4*)(feat + (size_t)src * FEAT))[c4];
        bf16x4 p; p[0] = tobf(v.x); p[1] = tobf(v.y); p[2] = tobf(v.z); p[3] = tobf(v.w);
        *(bf16x4*)(A + r * ASTRIDE + c4 * 4) = p;
    }
    if (tid < ROWS) {
        int grow = rowsBase + tid; if (grow > N_SUB - 1) grow = N_SUB - 1;
        ADJ[tid] = adj[grow];
    }
    __syncthreads();

    const int wave = tid >> 5;
    const int lane = tid & 31;
    const int half = lane >> 4;
    const int l16  = lane & 15;
    const int m0   = wave * 16;   // wave's 16-row M strip

    // ---- Stage 2: GEMM1  sub_xw(128x64) = A(128x512) @ W12 ----
    {
        f32x8 acc[4];
#pragma unroll
        for (int nt = 0; nt < 4; ++nt)
#pragma unroll
            for (int i = 0; i < 8; ++i) acc[nt][i] = 0.f;
        for (int kb = 0; kb < FEAT; kb += 32) {
            const __bf16* ar = A + (m0 + l16) * ASTRIDE + kb + half * 8;
            bf16x16 af = mk_frag(ar, ar + 16);
#pragma unroll
            for (int nt = 0; nt < 4; ++nt) {
                const __bf16* br = W12T + (nt * 16 + l16) * FEAT + kb + half * 8;
                acc[nt] = wmma_bf(af, mk_frag(br, br + 16), acc[nt]);
            }
        }
#pragma unroll
        for (int nt = 0; nt < 4; ++nt)
#pragma unroll
            for (int r = 0; r < 8; ++r)
                S[(m0 + half * 8 + r) * SSTRIDE + nt * 16 + l16] = tobf(acc[nt][r]);
    }
    __syncthreads();

    // ---- Stage 3: GEMM2  h1 = leaky(sub_xw @ W1[64:128] + adj*W1[192] + c1), bf16 into A ----
    {
        const __bf16* a0p = S + (m0 + l16) * SSTRIDE + 0  + half * 8;
        const __bf16* a1p = S + (m0 + l16) * SSTRIDE + 32 + half * 8;
        bf16x16 a0 = mk_frag(a0p, a0p + 16);
        bf16x16 a1 = mk_frag(a1p, a1p + 16);
        for (int nt = 0; nt < 32; ++nt) {
            f32x8 acc;
#pragma unroll
            for (int i = 0; i < 8; ++i) acc[i] = 0.f;
            const __bf16* b0 = W1sT + (nt * 16 + l16) * 64 + 0  + half * 8;
            const __bf16* b1f = W1sT + (nt * 16 + l16) * 64 + 32 + half * 8;
            acc = wmma_bf(a0, mk_frag(b0, b0 + 16), acc);
            acc = wmma_bf(a1, mk_frag(b1f, b1f + 16), acc);
            int col = nt * 16 + l16;
            float c1v = c1[col];
            float w1a = W1[192 * 512 + col];
#pragma unroll
            for (int r = 0; r < 8; ++r) {
                int rl = m0 + half * 8 + r;
                float h = leaky(acc[r] + ADJ[rl] * w1a + c1v);
                A[rl * ASTRIDE + col] = tobf(h);
            }
        }
    }
    __syncthreads();

    // ---- Stage 4: GEMM3  h2 = h1(128x512) @ W2(512x32); logits partials ----
    {
        f32x8 acc[2];
#pragma unroll
        for (int nt = 0; nt < 2; ++nt)
#pragma unroll
            for (int i = 0; i < 8; ++i) acc[nt][i] = 0.f;
        for (int kb = 0; kb < FEAT; kb += 32) {
            const __bf16* ar = A + (m0 + l16) * ASTRIDE + kb + half * 8;
            bf16x16 af = mk_frag(ar, ar + 16);
#pragma unroll
            for (int nt = 0; nt < 2; ++nt) {
                const __bf16* br = W2T + (nt * 16 + l16) * FEAT + kb + half * 8;
                acc[nt] = wmma_bf(af, mk_frag(br, br + 16), acc[nt]);
            }
        }
        float w3a = W3[l16],      w3b = W3[16 + l16];
        float b2a = b2[l16],      b2b = b2[16 + l16];
#pragma unroll
        for (int r = 0; r < 8; ++r) {
            float s = leaky(acc[0][r] + b2a) * w3a + leaky(acc[1][r] + b2b) * w3b;
            RED[tid * 8 + r] = s;  // [wave][lane][r]
        }
    }
    __syncthreads();

    // ---- final per-row reduction over 16 lanes -> logits ----
    if (tid < ROWS) {
        int w = tid >> 4, ri = tid & 15;
        int h2 = ri >> 3, r = ri & 7;
        const float* base = RED + ((w * 32 + h2 * 16) * 8) + r;
        float sum = 0.f;
#pragma unroll
        for (int j = 0; j < 16; ++j) sum += base[j * 8];
        int grow = rowsBase + tid;
        if (grow < N_SUB) logits[grow] = sum + b3[0];
    }
}

// ---------------- Gumbel top-k: 64 sequential softmax passes, multi-block ----------------
__device__ __forceinline__ unsigned okey(float f) {
    unsigned u = __float_as_uint(f);
    return (u & 0x80000000u) ? ~u : (u | 0x80000000u);
}
__device__ __forceinline__ float okey_inv(unsigned k) {
    unsigned u = (k & 0x80000000u) ? (k & 0x7FFFFFFFu) : ~k;
    return __uint_as_float(u);
}

__device__ __forceinline__ void gbar(unsigned* c, unsigned goal) {
    __syncthreads();
    __threadfence();
    if (threadIdx.x == 0) {
        atomicAdd(c, 1u);
        while (atomicAdd(c, 0u) < goal) { __builtin_amdgcn_s_sleep(8); }
    }
    __syncthreads();
    __threadfence();
}

extern "C" __global__ void __launch_bounds__(256)
topk_kernel(float* __restrict__ l, float* __restrict__ out,
            unsigned long long* __restrict__ maxkey, float* __restrict__ sumexp,
            unsigned* __restrict__ cnt) {
    __shared__ float rm[256];
    __shared__ int   rix[256];
    __shared__ float rs[256];
    const int tid = threadIdx.x;
    const int gtid = blockIdx.x * 256 + tid;
    const int gstride = GBLK * 256;
    const float inv = 100.f;  // 1/(0.01*tau)

    for (int t = 0; t < 64; ++t) {
        // A: global max/argmax of masked logits
        float m = -3.4e38f; int mi = 0;
        for (int i = gtid; i < N_SUB; i += gstride) {
            float v = l[i];
            if (v > m) { m = v; mi = i; }
        }
        rm[tid] = m; rix[tid] = mi;
        __syncthreads();
        for (int s = 128; s > 0; s >>= 1) {
            if (tid < s) {
                float vo = rm[tid + s]; int io = rix[tid + s];
                if (vo > rm[tid] || (vo == rm[tid] && io < rix[tid])) { rm[tid] = vo; rix[tid] = io; }
            }
            __syncthreads();
        }
        if (tid == 0) {
            unsigned long long key =
                ((unsigned long long)okey(rm[0]) << 32) | (unsigned)(~(unsigned)rix[0]);
            atomicMax(&maxkey[t], key);
        }
        gbar(&cnt[t * 4 + 0], GBLK);
        unsigned long long key = maxkey[t];
        float gm = okey_inv((unsigned)(key >> 32));
        int gidx = (int)(~(unsigned)key);

        // B: sum of exp
        float ps = 0.f;
        for (int i = gtid; i < N_SUB; i += gstride) ps += __expf((l[i] - gm) * inv);
        rs[tid] = ps;
        __syncthreads();
        for (int s = 128; s > 0; s >>= 1) {
            if (tid < s) rs[tid] += rs[tid + s];
            __syncthreads();
        }
        if (tid == 0) atomicAdd(&sumexp[t], rs[0]);
        gbar(&cnt[t * 4 + 1], GBLK);
        float rinv = 1.f / sumexp[t];

        // C: disc += softmax
        for (int i = gtid; i < N_SUB; i += gstride)
            out[i] += __expf((l[i] - gm) * inv) * rinv;
        gbar(&cnt[t * 4 + 2], GBLK);

        // mask the argmax for the next pass (after everyone consumed this pass)
        if (t < 63) {
            if (gtid == 0) l[gidx] -= 9999.f;
            gbar(&cnt[t * 4 + 3], GBLK);
        }
    }
}

// ---------------- launch ----------------
extern "C" void kernel_launch(void* const* d_in, const int* in_sizes, int n_in,
                              void* d_out, int out_size, void* d_ws, size_t ws_size,
                              hipStream_t stream) {
    (void)in_sizes; (void)n_in; (void)out_size; (void)ws_size;
    const int*   target  = (const int*)d_in[1];
    const int*   subn    = (const int*)d_in[2];
    const float* feat    = (const float*)d_in[3];
    const float* adj     = (const float*)d_in[4];
    const float* wlabel  = (const float*)d_in[5];
    const float* wsec    = (const float*)d_in[6];
    const float* weight1 = (const float*)d_in[7];
    const float* weight2 = (const float*)d_in[8];
    const float* W1      = (const float*)d_in[9];
    const float* b1      = (const float*)d_in[10];
    const float* W2      = (const float*)d_in[11];
    const float* b2      = (const float*)d_in[12];
    const float* W3      = (const float*)d_in[13];
    const float* b3      = (const float*)d_in[14];

    char* ws = (char*)d_ws;
    float*  W12   = (float*)(ws + 0);          // 512*64 f32
    __bf16* W12T  = (__bf16*)(ws + 131072);    // [64][512] bf16
    __bf16* W1sT  = (__bf16*)(ws + 196608);    // [512][64] bf16
    __bf16* W2T   = (__bf16*)(ws + 262144);    // [32][512] bf16
    float*  c1    = (float*)(ws + 294912);     // 512 f32
    float*  logit = (float*)(ws + 296960);     // 150000 f32
    char*   ctrl  = ws + 897024;               // topk control
    unsigned long long* maxkey = (unsigned long long*)ctrl;  // 64 u64
    float*    sumexp = (float*)(ctrl + 512);                 // 64 f32
    unsigned* cnt    = (unsigned*)(ctrl + 768);              // 256 u32

    float* out = (float*)d_out;

    prep_w12 <<<128, 256, 0, stream>>>(weight1, weight2, W12, W12T);
    prep_conv<<<192, 256, 0, stream>>>(W1, W2, W1sT, W2T);
    prep_const<<<1, 512, 0, stream>>>(feat, target, W12, wlabel, wsec, W1, b1, c1);
    init_out <<<(N_SUB + 255) / 256, 256, 0, stream>>>(subn, out, (unsigned*)ctrl);

    size_t smem = (size_t)ROWS * ASTRIDE * 2 + (size_t)ROWS * SSTRIDE * 2 + (size_t)ROWS * 4;
    fused_main<<<(N_SUB + ROWS - 1) / ROWS, 256, smem, stream>>>(
        subn, feat, adj, W12T, W1sT, W2T, c1, W1, b2, W3, b3, logit);

    topk_kernel<<<GBLK, 256, 0, stream>>>(logit, out, maxkey, sumexp, cnt);
}
